// MoELayer_3736621547981
// MI455X (gfx1250) — matrix operand order repaired and checked by reading, not compiled, not run on hardware
//
#include <hip/hip_runtime.h>

#define H 1024
#define F 4096
#define E 8
#define T 4096            // B*S = 4*1024
#define FCHUNK 512
#define NFCHUNK (F / FCHUNK)

typedef __attribute__((ext_vector_type(16))) __bf16        v16bf;
typedef __attribute__((ext_vector_type(8)))  float         v8f;
typedef __attribute__((ext_vector_type(8)))  unsigned int  v8u;

// ---- bf16 packing helpers (round-to-nearest-even-ish) ----
__device__ __forceinline__ unsigned pack2_bf16(float a, float b) {
    unsigned ua = __builtin_bit_cast(unsigned, a);
    unsigned ub = __builtin_bit_cast(unsigned, b);
    ua += 0x7FFFu + ((ua >> 16) & 1u);
    ub += 0x7FFFu + ((ub >> 16) & 1u);
    return (ub & 0xFFFF0000u) | (ua >> 16);
}
__device__ __forceinline__ unsigned short f2bf(float a) {
    unsigned ua = __builtin_bit_cast(unsigned, a);
    ua += 0x7FFFu + ((ua >> 16) & 1u);
    return (unsigned short)(ua >> 16);
}

// A-fragment (16x32 bf16) from an LDS row: lane holds row M=lane&15;
// lanes 0-15 hold K k0..k0+15, lanes 16-31 get +16 (caller passes koff).
__device__ __forceinline__ v16bf lds_load_a(const unsigned short* rowp, int koff) {
    uint4 a = *(const uint4*)(rowp + koff);
    uint4 b = *(const uint4*)(rowp + koff + 8);
    v8u u = {a.x, a.y, a.z, a.w, b.x, b.y, b.z, b.w};
    return __builtin_bit_cast(v16bf, u);
}

// B-fragment (32x16 bf16) from K-contiguous (transposed) bf16 weights:
// lane holds column N=lane&15; lanes 0-15 take K k0..k0+15, lanes 16-31 K k0+16..k0+31
// (caller bakes the +16 into the pointer). 32 contiguous bytes -> 2x b128.
__device__ __forceinline__ v16bf load_b16(const unsigned short* p) {
    uint4 a = *(const uint4*)(p);
    uint4 b = *(const uint4*)(p + 8);
    v8u u = {a.x, a.y, a.z, a.w, b.x, b.y, b.z, b.w};
    return __builtin_bit_cast(v16bf, u);
}

// ---------------- kernel 0: zero per-expert counters ----------------
__global__ void moe_init(int* __restrict__ counts) {
    if (threadIdx.x < E) counts[threadIdx.x] = 0;
}

// ------- kernel T: fp32 [e][R][C] -> bf16 [e][C][R] (transpose+convert) -------
// Block = 256 threads, handles one 64(R) x 32(C) tile. Reads coalesced along C,
// writes packed 2xbf16 uints coalesced along R.
__global__ __launch_bounds__(256) void moe_transpose(
    const float* __restrict__ src, unsigned short* __restrict__ dst,
    int R, int C) {
    __shared__ unsigned short tile[32][65];   // [c][r], padded
    const int tilesC   = C / 32;
    const int tilesExp = (R / 64) * tilesC;
    const int e   = blockIdx.x / tilesExp;
    const int tid = blockIdx.x % tilesExp;
    const int tr  = tid / tilesC;
    const int tc  = tid % tilesC;
    const int r0 = tr * 64, c0 = tc * 32;
    const int tx = threadIdx.x & 31;
    const int ty = threadIdx.x >> 5;          // 0..7
    const float* s   = src + (size_t)e * R * C;
    unsigned short* d = dst + (size_t)e * R * C;
#pragma unroll
    for (int i = 0; i < 8; ++i) {
        int r = ty + i * 8;                   // 0..63
        tile[tx][r] = f2bf(s[(size_t)(r0 + r) * C + c0 + tx]);
    }
    __syncthreads();
#pragma unroll
    for (int i = 0; i < 4; ++i) {
        int c = ty + i * 8;                   // 0..31
        unsigned lo = tile[c][2 * tx];
        unsigned hi = tile[c][2 * tx + 1];
        *(unsigned*)(d + (size_t)(c0 + c) * R + r0 + 2 * tx) = (hi << 16) | lo;
    }
}

// ---------------- kernel 1: router (one wave per token) ----------------
__global__ __launch_bounds__(256) void moe_router(
    const float* __restrict__ x, const float* __restrict__ rw,
    const float* __restrict__ rb, float* __restrict__ gates,
    int* __restrict__ counts, int* __restrict__ lists) {
    const int lane = threadIdx.x & 31;
    const int wv   = threadIdx.x >> 5;
    const int t    = blockIdx.x * 8 + wv;
    const float* xr = x + (size_t)t * H;

    float l[E];
#pragma unroll
    for (int e = 0; e < E; ++e) l[e] = 0.f;
    for (int h0 = lane; h0 < H; h0 += 32) {
        float xv = xr[h0];
        const float4* wr = (const float4*)(rw + (size_t)h0 * E);
        float4 w0 = wr[0], w1v = wr[1];
        l[0] += xv * w0.x;  l[1] += xv * w0.y;  l[2] += xv * w0.z;  l[3] += xv * w0.w;
        l[4] += xv * w1v.x; l[5] += xv * w1v.y; l[6] += xv * w1v.z; l[7] += xv * w1v.w;
    }
#pragma unroll
    for (int e = 0; e < E; ++e) {
#pragma unroll
        for (int off = 16; off > 0; off >>= 1) l[e] += __shfl_xor(l[e], off, 32);
    }
    if (lane == 0) {
#pragma unroll
        for (int e = 0; e < E; ++e) l[e] += rb[e];
        int i1 = 0;
#pragma unroll
        for (int e = 1; e < E; ++e) if (l[e] > l[i1]) i1 = e;
        int i2 = (i1 == 0) ? 1 : 0;
#pragma unroll
        for (int e = 0; e < E; ++e) if (e != i1 && l[e] > l[i2]) i2 = e;
        // renormalized top-2 softmax: full denominator cancels
        float p2  = __expf(l[i2] - l[i1]);
        float inv = 1.f / (1.f + p2);
        gates[2 * t]     = inv;
        gates[2 * t + 1] = p2 * inv;
        int q1 = atomicAdd(&counts[i1], 1);
        lists[i1 * T + q1] = (t << 1);
        int q2 = atomicAdd(&counts[i2], 1);
        lists[i2 * T + q2] = (t << 1) | 1;
    }
}

// ---------------- kernel 2: fused expert MLP via bf16 WMMA ----------------
// grid = E*256 blocks (expert-major: one expert's bf16 weights = 16.8MB, L2-resident),
// 512 threads = 16 waves. Block handles 16 assignments; wave owns 64 output cols.
__global__ __launch_bounds__(512) void moe_expert(
    const float* __restrict__ x,
    const unsigned short* __restrict__ w1t,   // bf16 [E][F][H]
    const float* __restrict__ b1,
    const unsigned short* __restrict__ w2t,   // bf16 [E][H][F]
    const float* __restrict__ b2,
    const int* __restrict__ counts, const int* __restrict__ lists,
    float* __restrict__ ypair) {
    __shared__ unsigned short sX[16][1032];   // 16 tokens x H bf16 (+pad)
    __shared__ unsigned short sH[16][520];    // 16 tokens x FCHUNK bf16 (+pad)
    __shared__ int sEntry[16];
    __shared__ int sTok[16];

    const int e    = blockIdx.x >> 8;
    const int tile = blockIdx.x & 255;
    const int cnt  = counts[e];
    if (tile * 16 >= cnt) return;

    if (threadIdx.x < 16) {
        int gi  = tile * 16 + threadIdx.x;
        int ent = (gi < cnt) ? lists[e * T + gi] : -1;
        sEntry[threadIdx.x] = ent;
        sTok[threadIdx.x]   = (ent >= 0) ? (ent >> 1) : 0;
    }
    __syncthreads();

    const int lane = threadIdx.x & 31;
    const int wv   = threadIdx.x >> 5;     // 0..15
    const int colL = lane & 15;
    const int hiH  = (lane >> 4) & 1;
    const int kofsA = hiH * 16;            // A-fragment K offset for this half
    const int kofsB = hiH * 16;            // B-fragment K offset for this half

    // ---- stage X tile (fp32 -> bf16) into LDS, coalesced float4 loads ----
    {
        const int row = wv;
        const float* xr = x + (size_t)sTok[row] * H;
        unsigned short* rp = &sX[row][0];
        const int c0 = lane * 32;
#pragma unroll
        for (int j = 0; j < 32; j += 4) {
            float4 v = *(const float4*)(xr + c0 + j);
            *(unsigned*)(rp + c0 + j)     = pack2_bf16(v.x, v.y);
            *(unsigned*)(rp + c0 + j + 2) = pack2_bf16(v.z, v.w);
        }
    }
    __syncthreads();

    v8f acc[4];
#pragma unroll
    for (int c = 0; c < 4; ++c) acc[c] = (v8f){0.f,0.f,0.f,0.f,0.f,0.f,0.f,0.f};

    const unsigned short* w1e = w1t + (size_t)e * F * H;   // [F][H]: K contiguous
    const unsigned short* w2e = w2t + (size_t)e * H * F;   // [H][F]: K contiguous
    const unsigned short* aRowX = &sX[colL][0];
    const unsigned short* aRowH = &sH[colL][0];

    for (int fc = 0; fc < NFCHUNK; ++fc) {
        // ---- GEMM1: h[16 x FCHUNK] = relu(X @ W1[:, chunk] + b1) ----
#pragma unroll
        for (int c2 = 0; c2 < 2; ++c2) {
            const int fg = fc * FCHUNK + wv * 32 + c2 * 16 + colL;
            const unsigned short* bcol = w1e + (size_t)fg * H + kofsB;
            v8f hacc = (v8f){0.f,0.f,0.f,0.f,0.f,0.f,0.f,0.f};
#pragma unroll 4
            for (int ks = 0; ks < 32; ++ks) {           // K = H = 1024
                const int k0 = ks * 32;
                __builtin_prefetch(bcol + k0 + 256, 0, 1);
                v16bf A  = lds_load_a(aRowX, k0 + kofsA);
                v16bf Bv = load_b16(bcol + k0);
                hacc = __builtin_amdgcn_wmma_f32_16x16x32_bf16(
                    false, A, false, Bv, (short)0, hacc, false, false);
            }
            const float b1v = b1[e * F + fg];
            const int cl = wv * 32 + c2 * 16 + colL;
#pragma unroll
            for (int i = 0; i < 8; ++i) {
                float hv = hacc[i] + b1v;
                hv = hv > 0.f ? hv : 0.f;
                sH[i + hiH * 8][cl] = f2bf(hv);
            }
        }
        __syncthreads();

        // ---- GEMM2: acc[16 x 64 per wave] += h_chunk @ W2[chunk, :] ----
#pragma unroll 2
        for (int ks = 0; ks < 16; ++ks) {               // K = FCHUNK = 512
            const int k0 = ks * 32;
            v16bf A = lds_load_a(aRowH, k0 + kofsA);
#pragma unroll
            for (int c = 0; c < 4; ++c) {
                const int n = wv * 64 + c * 16 + colL;
                const unsigned short* bp =
                    w2e + (size_t)n * F + fc * FCHUNK + k0 + kofsB;
                v16bf Bv = load_b16(bp);
                acc[c] = __builtin_amdgcn_wmma_f32_16x16x32_bf16(
                    false, A, false, Bv, (short)0, acc[c], false, false);
            }
        }
        __syncthreads();
    }

    // ---- epilogue: y = acc + b2 scattered to y_pair[t][slot][:] ----
#pragma unroll
    for (int c = 0; c < 4; ++c) {
        const int n = wv * 64 + c * 16 + colL;
        const float b2v = b2[e * H + n];
#pragma unroll
        for (int i = 0; i < 8; ++i) {
            const int m   = i + hiH * 8;
            const int ent = sEntry[m];
            if (ent >= 0) ypair[(size_t)ent * H + n] = acc[c][i] + b2v;
        }
    }
}

// ---------------- kernel 3: gated combine ----------------
__global__ __launch_bounds__(256) void moe_combine(
    const float* __restrict__ gates, const float* __restrict__ ypair,
    float* __restrict__ out) {
    const int idx = blockIdx.x * 256 + threadIdx.x;
    const int t   = idx >> 8;            // (idx*4) / H
    const int col = (idx & 255) * 4;
    const float g0 = gates[2 * t], g1 = gates[2 * t + 1];
    const float4 y0 = *(const float4*)(ypair + (size_t)(2 * t)     * H + col);
    const float4 y1 = *(const float4*)(ypair + (size_t)(2 * t + 1) * H + col);
    float4 o;
    o.x = g0 * y0.x + g1 * y1.x;
    o.y = g0 * y0.y + g1 * y1.y;
    o.z = g0 * y0.z + g1 * y1.z;
    o.w = g0 * y0.w + g1 * y1.w;
    *(float4*)(out + (size_t)t * H + col) = o;
}

extern "C" void kernel_launch(void* const* d_in, const int* in_sizes, int n_in,
                              void* d_out, int out_size, void* d_ws, size_t ws_size,
                              hipStream_t stream) {
    const float* x  = (const float*)d_in[0];
    const float* rw = (const float*)d_in[1];
    const float* rb = (const float*)d_in[2];
    const float* w1 = (const float*)d_in[3];
    const float* b1 = (const float*)d_in[4];
    const float* w2 = (const float*)d_in[5];
    const float* b2 = (const float*)d_in[6];
    float* out = (float*)d_out;

    char* ws = (char*)d_ws;
    int*   counts = (int*)ws;                                   // E ints
    int*   lists  = (int*)(ws + 4096);                          // E*T ints  (128 KB)
    float* gates  = (float*)(ws + 256 * 1024);                  // T*2 floats (32 KB)
    float* ypair  = (float*)(ws + 512 * 1024);                  // T*2*H floats (33.5 MB)
    unsigned short* w1t = (unsigned short*)(ws + (size_t)48  * 1024 * 1024); // 64 MB
    unsigned short* w2t = (unsigned short*)(ws + (size_t)112 * 1024 * 1024); // 64 MB

    moe_init     <<<1, 32, 0, stream>>>(counts);
    // w1 [E][H][F] -> w1t bf16 [E][F][H];  w2 [E][F][H] -> w2t bf16 [E][H][F]
    moe_transpose<<<E * (H / 64) * (F / 32), 256, 0, stream>>>(w1, w1t, H, F);
    moe_transpose<<<E * (F / 64) * (H / 32), 256, 0, stream>>>(w2, w2t, F, H);
    moe_router   <<<T / 8, 256, 0, stream>>>(x, rw, rb, gates, counts, lists);
    moe_expert   <<<E * 256, 512, 0, stream>>>(x, w1t, b1, w2t, b2, counts, lists, ypair);
    moe_combine  <<<(T * H) / (4 * 256), 256, 0, stream>>>(gates, ypair, out);
}
